// Model_59339268162190
// MI455X (gfx1250) — matrix operand order repaired and checked by reading, not compiled, not run on hardware
//
#include <hip/hip_runtime.h>
#include <hip/hip_bf16.h>
#include <math.h>

// ---------------------------------------------------------------------------
// MI455X (gfx1250) implementation of the piecewise-resonance synth forward.
// Matrix-heavy stages use v_wmma_f32_16x16x32_f16 (wave32 WMMA).
// Long convolution FFT: radix-4 Stockham (8 passes).
// GEMM is templated on the (compile-time) B stride so B-fragment loads lower
// to immediate-offset global_load_b32 clauses instead of 64-bit add chains.
// ---------------------------------------------------------------------------

typedef __attribute__((ext_vector_type(16))) _Float16 v16h;
typedef __attribute__((ext_vector_type(8)))  float    v8f;

#define NRES   32768
#define NFFT   65536
#define NQ     16384           // NFFT/4
#define TWOPI_F 6.28318530717958647692f

// ===========================================================================
// Generic f16 WMMA GEMM:  C[M,N] = A[M,K] * B[K,N]  (A,B f32 in memory,
// converted to f16 in the fragment loaders; f32 accumulate).
// MODE 0: C[m*ldc+n] = acc + bias[n]
// MODE 1: res path — multiply by linearly-interpolated decay(b = m>>2, t = n)
//         and scatter into the complex FFT buffer (re part, row m).
// Block = 4 waves (32x4); each wave owns one 16x16 C tile. All guards are
// wave-uniform so EXEC is all-ones at every WMMA. K-loop unrolled x2.
// ===========================================================================
__device__ __forceinline__ v16h load_fragA(const float* __restrict__ A, int lda,
                                           int r, int k, int half)
{
    // A fragment (contiguous k, same order as B): half 0 -> K 0..15, half 1 -> K 16..31
    const float* ap = A + (size_t)r * lda + (k + half * 16);
    v16h af;
#pragma unroll
    for (int i = 0; i < 8; ++i) {
        af[i]     = (_Float16)ap[i];
        af[8 + i] = (_Float16)ap[8 + i];
    }
    return af;
}

template <int LDB>
__device__ __forceinline__ v16h load_fragB(const float* __restrict__ Bm,
                                           int k, int c, int half)
{
    // B fragment (contiguous k, same order as A): half 0 -> K 0..15, half 1 -> K 16..31 (column c per lane)
    // LDB is a compile-time constant -> i*LDB folds into immediate offsets.
    const float* bp = Bm + (size_t)(k + half * 16) * LDB + c;
    v16h bf;
#pragma unroll
    for (int i = 0; i < 16; ++i)
        bf[i] = (_Float16)bp[i * LDB];
    return bf;
}

template <int LDB, int MODE>
__global__ void __launch_bounds__(128)
k_gemm(const float* __restrict__ A, int lda,
       const float* __restrict__ Bm,
       float* __restrict__ C, int ldc, const float* __restrict__ bias,
       int M, int N, int K,
       const float* __restrict__ decay, float* __restrict__ cplx)
{
    const int lane = threadIdx.x;                       // 0..31
    const int n0 = (blockIdx.x * 4 + threadIdx.y) * 16; // wave-uniform
    const int m0 = blockIdx.y * 16;
    if (n0 >= N || m0 >= M) return;                     // wave-uniform exit

    const int half = lane >> 4;           // 0 or 1
    const int r = m0 + (lane & 15);       // A row for this lane
    const int c = n0 + (lane & 15);       // B/C column for this lane

    v8f acc = {0.f, 0.f, 0.f, 0.f, 0.f, 0.f, 0.f, 0.f};

    for (int k0 = 0; k0 < K; k0 += 64) {
        // prefetch the next B panel (streaming path: atoms) -> global_prefetch_b8
        if (k0 + 64 < K)
            __builtin_prefetch(Bm + (size_t)(k0 + 64 + half * 16) * LDB + c, 0, 1);

        v16h af = load_fragA(A, lda, r, k0, half);
        v16h bf = load_fragB<LDB>(Bm, k0, c, half);
        acc = __builtin_amdgcn_wmma_f32_16x16x32_f16(
            false, af, false, bf, (short)0, acc, false, false);

        if (k0 + 32 < K) {                              // wave-uniform
            v16h af2 = load_fragA(A, lda, r, k0 + 32, half);
            v16h bf2 = load_fragB<LDB>(Bm, k0 + 32, c, half);
            acc = __builtin_amdgcn_wmma_f32_16x16x32_f16(
                false, af2, false, bf2, (short)0, acc, false, false);
        }
    }

    // D layout: element i -> (row m0 + i + 8*half, col c)
    const int rr = m0 + half * 8;
    if (MODE == 0) {
        const float bv = bias ? bias[c] : 0.f;
#pragma unroll
        for (int i = 0; i < 8; ++i)
            C[(size_t)(rr + i) * ldc + c] = acc[i] + bv;
    } else {
        // decay interp: coords = (t+0.5)/256 - 0.5, clamped to [0,127]
        float tc = ((float)c + 0.5f) * (1.f / 256.f) - 0.5f;
        tc = fminf(fmaxf(tc, 0.f), 127.f);
        const int lo = (int)tc;
        const int hi = (lo + 1 > 127) ? 127 : lo + 1;
        const float w = tc - (float)lo;
#pragma unroll
        for (int i = 0; i < 8; ++i) {
            const int m = rr + i;                 // m = b*4 + p
            const float* dr = decay + (size_t)(m >> 2) * 128;
            const float dv = dr[lo] * (1.f - w) + dr[hi] * w;
            cplx[((size_t)m * NFFT + c) * 2] = acc[i] * dv;   // re; im stays 0
        }
    }
}

// ===========================================================================
// Small kernels
// ===========================================================================
__global__ void k_zero(float* __restrict__ p, size_t n)
{
    size_t i = (size_t)blockIdx.x * blockDim.x + threadIdx.x;
    const size_t stride = (size_t)gridDim.x * blockDim.x;
    for (; i < n; i += stride) p[i] = 0.f;
}

// decay envelope: d128[b,f] = exp(cumsum_f log(1e-12 + 0.02 + 0.931*sigmoid(.)))
__global__ void k_decay(const float* __restrict__ latent,
                        const float* __restrict__ dec_w,
                        const float* __restrict__ dec_b,
                        float* __restrict__ d128)
{
    const int b = blockIdx.x, f = threadIdx.x;     // 64 blocks x 128 threads
    const float* lp = latent + (size_t)b * 256;
    float s = dec_b[f];
    for (int l = 0; l < 256; ++l) s += lp[l] * dec_w[(size_t)l * 128 + f];
    const float sig = 1.f / (1.f + __expf(-s));
    const float d = 0.02f + sig * 0.931f;          // BASE_RES + d*RES_FACTOR
    __shared__ float lg[128];
    lg[f] = __logf(1e-12f + d);
    __syncthreads();
    float cs = 0.f;
    for (int j = 0; j <= f; ++j) cs += lg[j];
    d128[(size_t)b * 128 + f] = __expf(cs);
}

// row softmax over 2048 (sel logits, bias already added by GEMM)
__global__ void k_softmax2048(float* __restrict__ x)
{
    float* p = x + (size_t)blockIdx.x * 2048;
    __shared__ float red[256];
    const int tid = threadIdx.x;
    float m = -1e30f;
    for (int i = tid; i < 2048; i += 256) m = fmaxf(m, p[i]);
    red[tid] = m; __syncthreads();
    for (int o = 128; o > 0; o >>= 1) { if (tid < o) red[tid] = fmaxf(red[tid], red[tid + o]); __syncthreads(); }
    m = red[0]; __syncthreads();
    float s = 0.f;
    for (int i = tid; i < 2048; i += 256) { float e = __expf(p[i] - m); p[i] = e; s += e; }
    red[tid] = s; __syncthreads();
    for (int o = 128; o > 0; o >>= 1) { if (tid < o) red[tid] += red[tid + o]; __syncthreads(); }
    const float inv = 1.f / red[0];
    for (int i = tid; i < 2048; i += 256) p[i] *= inv;
}

// impulse -> complex buffer rows 256..319 (re only; rest is zero-filled)
__global__ void k_pack_imp(const float* __restrict__ imp, float* __restrict__ cplx)
{
    const int idx = blockIdx.x * 256 + threadIdx.x;
    if (idx >= 64 * 16384) return;
    const int b = idx >> 14, t = idx & 16383;
    cplx[(((size_t)(256 + b) << 16) + t) * 2] = imp[idx];
}

__device__ __forceinline__ float2 cmul(float2 a, float2 b)
{
    return make_float2(a.x * b.x - a.y * b.y, a.x * b.y + a.y * b.x);
}

// One Stockham radix-4 stage (natural-order autosort), forward sign.
// N = 65536 = 4^8 -> 8 stages. s = 4^st, n = N >> 2st, m = n/4.
// y[q+s(4p+k)] = W_n^{pk} * X_k(butterfly of x[q+s(p+j*m)], j=0..3)
__global__ void k_fft4_stage(const float2* __restrict__ in, float2* __restrict__ out,
                             int rows, int st)
{
    long long g = (long long)blockIdx.x * blockDim.x + threadIdx.x;
    if (g >= (long long)rows * NQ) return;
    const int row = (int)(g >> 14);
    const int t   = (int)(g & (NQ - 1));
    const int sh  = 2 * st;
    const int s   = 1 << sh;              // stride
    const int n   = NFFT >> sh;           // current transform length
    const int p   = t >> sh;              // p in [0, n/4)
    const int q   = t & (s - 1);

    const float2* x = in  + ((size_t)row << 16);
    float2*       y = out + ((size_t)row << 16);

    const int base = q + (p << sh);
    const float2 a = x[base];
    const float2 b = x[base + NQ];
    const float2 c = x[base + 2 * NQ];
    const float2 d = x[base + 3 * NQ];

    const float2 apc = make_float2(a.x + c.x, a.y + c.y);
    const float2 amc = make_float2(a.x - c.x, a.y - c.y);
    const float2 bpd = make_float2(b.x + d.x, b.y + d.y);
    const float2 bmd = make_float2(b.x - d.x, b.y - d.y);

    // forward (e^{-i}): X1 = amc - i*bmd, X3 = amc + i*bmd
    const float2 x0 = make_float2(apc.x + bpd.x, apc.y + bpd.y);
    const float2 x1 = make_float2(amc.x + bmd.y, amc.y - bmd.x);
    const float2 x2 = make_float2(apc.x - bpd.x, apc.y - bpd.y);
    const float2 x3 = make_float2(amc.x - bmd.y, amc.y + bmd.x);

    const float ang = -TWOPI_F * (float)p / (float)n;
    float sn, cs; __sincosf(ang, &sn, &cs);
    const float2 w1 = make_float2(cs, sn);
    const float2 w2 = cmul(w1, w1);
    const float2 w3 = cmul(w2, w1);

    const int ob = q + ((4 * p) << sh);
    y[ob]         = x0;
    y[ob + s]     = cmul(w1, x1);
    y[ob + 2 * s] = cmul(w2, x2);
    y[ob + 3 * s] = cmul(w3, x3);
}

// spectral product + conjugate (sets up inverse-FFT-via-forward-FFT trick)
__global__ void k_pointwise(const float2* __restrict__ in, float2* __restrict__ out)
{
    long long g = (long long)blockIdx.x * blockDim.x + threadIdx.x;
    if (g >= (long long)256 * NFFT) return;
    const int r = (int)(g >> 16);
    const int i = (int)(g & (NFFT - 1));
    const float2 X = in[((size_t)r << 16) + i];
    const float2 Y = in[((size_t)(256 + (r >> 2)) << 16) + i];
    out[((size_t)r << 16) + i] =
        make_float2(X.x * Y.x - X.y * Y.y, -(X.x * Y.y + X.y * Y.x));
}

// ConvUpsample (nearest x2 + conv k=3 pad=1) + leaky relu
__global__ void k_conv_up2(const float* __restrict__ in, float* __restrict__ out,
                           const float* __restrict__ w, const float* __restrict__ bias, int L)
{
    const int L2 = 2 * L;
    const int idx = blockIdx.x * blockDim.x + threadIdx.x;
    if (idx >= 64 * 128 * L2) return;
    const int t = idx % L2, co = (idx / L2) & 127, b = idx / (L2 * 128);
    const float* inb = in + (size_t)b * 128 * L;
    float acc = bias[co];
    for (int ci = 0; ci < 128; ++ci) {
        const float* wp = w + ((size_t)co * 128 + ci) * 3;
        const float* ip = inb + (size_t)ci * L;
#pragma unroll
        for (int d = 0; d < 3; ++d) {
            const int u = t + d - 1;
            if (u >= 0 && u < L2) acc += wp[d] * ip[u >> 1];
        }
    }
    out[idx] = acc > 0.f ? acc : 0.2f * acc;
}

// ConvTranspose1d(C,C,4,2,1) + leaky relu
__global__ void k_convtrans2(const float* __restrict__ in, float* __restrict__ out,
                             const float* __restrict__ w, const float* __restrict__ bias, int L)
{
    const int L2 = 2 * L;
    const int idx = blockIdx.x * blockDim.x + threadIdx.x;
    if (idx >= 64 * 128 * L2) return;
    const int t = idx % L2, co = (idx / L2) & 127, b = idx / (L2 * 128);
    const float* inb = in + (size_t)b * 128 * L;
    float acc = bias[co];
    for (int ci = 0; ci < 128; ++ci) {
        const float* wp = w + ((size_t)co * 128 + ci) * 4;
        const float* ip = inb + (size_t)ci * L;
#pragma unroll
        for (int k = 0; k < 4; ++k) {
            const int u = t + k - 2;
            if (u >= 0 && u < 2 * L - 1 && !(u & 1)) acc += wp[k] * ip[u >> 1];
        }
    }
    out[idx] = acc > 0.f ? acc : 0.2f * acc;
}

// training-mode BatchNorm1d over (batch, length), in place. One block/channel.
__global__ void k_bnorm(float* __restrict__ x, const float* __restrict__ g,
                        const float* __restrict__ bta, int L)
{
    const int c = blockIdx.x;                       // 128 channels
    const int tid = threadIdx.x;
    const int n = 64 * L;
    __shared__ float s1[256], s2[256];
    float sum = 0.f, sq = 0.f;
    for (int i = tid; i < n; i += 256) {
        const int b = i / L, t = i % L;
        const float v = x[((size_t)b * 128 + c) * L + t];
        sum += v; sq += v * v;
    }
    s1[tid] = sum; s2[tid] = sq; __syncthreads();
    for (int o = 128; o > 0; o >>= 1) {
        if (tid < o) { s1[tid] += s1[tid + o]; s2[tid] += s2[tid + o]; }
        __syncthreads();
    }
    const float m = s1[0] / n;
    const float v = s2[0] / n - m * m;
    const float inv = rsqrtf(v + 1e-5f);
    const float gg = g[c], bb = bta[c];
    for (int i = tid; i < n; i += 256) {
        const int b = i / L, t = i % L;
        const size_t id = ((size_t)b * 128 + c) * L + t;
        x[id] = (x[id] - m) * inv * gg + bb;
    }
}

// output conv k=3 pad=1, 128 -> Cout channels, length 128 (no activation)
__global__ void k_conv_out(const float* __restrict__ in, float* __restrict__ out,
                           const float* __restrict__ w, const float* __restrict__ bias, int Cout)
{
    const int idx = blockIdx.x * blockDim.x + threadIdx.x;
    if (idx >= 64 * Cout * 128) return;
    const int t = idx & 127, co = (idx >> 7) % Cout, b = idx / (128 * Cout);
    const float* inb = in + (size_t)b * 128 * 128;
    float acc = bias[co];
    for (int ci = 0; ci < 128; ++ci) {
        const float* wp = w + ((size_t)co * 128 + ci) * 3;
        const float* ip = inb + (size_t)ci * 128;
#pragma unroll
        for (int d = 0; d < 3; ++d) {
            const int u = t + d - 1;
            if (u >= 0 && u < 128) acc += wp[d] * ip[u];
        }
    }
    out[idx] = acc;
}

// filt[b,f,k] = lin_interp_k(sigmoid(filt32[b,:,f]), 32 -> 257)
__global__ void k_filt(const float* __restrict__ f32buf, float* __restrict__ filt)
{
    const int idx = blockIdx.x * blockDim.x + threadIdx.x;
    if (idx >= 64 * 128 * 257) return;
    const int kbin = idx % 257, f = (idx / 257) & 127, b = idx / (257 * 128);
    float c = (kbin + 0.5f) * (32.f / 257.f) - 0.5f;
    c = fminf(fmaxf(c, 0.f), 31.f);
    const int lo = (int)c;
    const int hi = (lo + 1 > 31) ? 31 : lo + 1;
    const float w = c - (float)lo;
    const float vlo = f32buf[((size_t)b * 32 + lo) * 128 + f];
    const float vhi = f32buf[((size_t)b * 32 + hi) * 128 + f];
    const float slo = 1.f / (1.f + __expf(-vlo));
    const float shi = 1.f / (1.f + __expf(-vhi));
    filt[idx] = slo * (1.f - w) + shi * w;
}

// combine: mix(avgpool9(interp(mx128)), softmax over 4) dot Re(ifft)/NFFT
__global__ void k_combine(const float2* __restrict__ F, const float* __restrict__ mx128,
                          float* __restrict__ finalc)
{
    const int idx = blockIdx.x * blockDim.x + threadIdx.x;
    if (idx >= 64 * NRES) return;
    const int b = idx >> 15, t = idx & (NRES - 1);
    float mv[4];
#pragma unroll
    for (int c = 0; c < 4; ++c) {
        const float* mp = mx128 + (size_t)(b * 4 + c) * 128;
        float s = 0.f;
#pragma unroll
        for (int dt = -4; dt <= 4; ++dt) {
            const int u = t + dt;
            if (u >= 0 && u < NRES) {
                float cc = ((float)u + 0.5f) * (1.f / 256.f) - 0.5f;
                cc = fminf(fmaxf(cc, 0.f), 127.f);
                const int lo = (int)cc;
                const int hi = (lo + 1 > 127) ? 127 : lo + 1;
                const float w = cc - (float)lo;
                s += mp[lo] * (1.f - w) + mp[hi] * w;
            }
        }
        mv[c] = s * (1.f / 9.f);
    }
    float mmax = fmaxf(fmaxf(mv[0], mv[1]), fmaxf(mv[2], mv[3]));
    float msum = 0.f;
#pragma unroll
    for (int c = 0; c < 4; ++c) { mv[c] = __expf(mv[c] - mmax); msum += mv[c]; }
    const float inv = 1.f / msum;
    float acc = 0.f;
#pragma unroll
    for (int p = 0; p < 4; ++p)
        acc += mv[p] * inv * F[((size_t)(b * 4 + p) << 16) + t].x;
    finalc[idx] = acc * (1.f / (float)NFFT);
}

// windowed frames (hann) + per-frame unit norm. One block per frame.
__global__ void k_window(const float* __restrict__ finalc, float* __restrict__ framesN)
{
    const int frame = blockIdx.x;          // 8192 = b*128 + f
    const int b = frame >> 7, f = frame & 127;
    const int tid = threadIdx.x;           // 256
    __shared__ float buf[512];
    __shared__ float red[256];
    for (int i = tid; i < 512; i += 256) {
        const int gt = f * 256 + i;
        const float v = (gt < NRES) ? finalc[((size_t)b << 15) + gt] : 0.f;
        const float h = 0.5f * (1.f - __cosf(TWOPI_F * (float)i / 512.f));
        buf[i] = v * h;
    }
    __syncthreads();
    red[tid] = buf[tid] * buf[tid] + buf[tid + 256] * buf[tid + 256];
    __syncthreads();
    for (int o = 128; o > 0; o >>= 1) { if (tid < o) red[tid] += red[tid + o]; __syncthreads(); }
    const float inv = 1.f / (sqrtf(red[0]) + 1e-8f);
    framesN[(size_t)frame * 512 + tid]       = buf[tid] * inv;
    framesN[(size_t)frame * 512 + tid + 256] = buf[tid + 256] * inv;
}

// DFT matrices: Wf (512 x 544): cos | -sin | 0-pad. Wi (544 x 512): irfft rows.
__global__ void k_gen_dft(float* __restrict__ Wf, float* __restrict__ Wi)
{
    const int gid = blockIdx.x * 256 + threadIdx.x;
    if (gid < 512 * 544) {
        const int i = gid / 544, j = gid % 544;
        float v = 0.f;
        if (j < 257) {
            const int ph = (i * j) & 511;
            v = __cosf(TWOPI_F * (float)ph / 512.f);
        } else if (j < 514) {
            const int jj = j - 257;
            const int ph = (i * jj) & 511;
            v = -__sinf(TWOPI_F * (float)ph / 512.f);
        }
        Wf[gid] = v;
    } else {
        const int g = gid - 512 * 544;
        if (g < 544 * 512) {
            const int k = g / 512, i = g % 512;
            float v = 0.f;
            if (k < 257) {
                const float cs = (k == 0 || k == 256) ? 1.f : 2.f;
                const int ph = (i * k) & 511;
                v = cs * __cosf(TWOPI_F * (float)ph / 512.f) * (1.f / 512.f);
            } else if (k < 514) {
                const int jj = k - 257;
                const float cs = (jj == 0 || jj == 256) ? 1.f : 2.f;
                const int ph = (i * jj) & 511;
                v = -cs * __sinf(TWOPI_F * (float)ph / 512.f) * (1.f / 512.f);
            }
            Wi[g] = v;
        }
    }
}

// spec *= filt (per-bin real gain; re cols 0..256, im cols 257..513; ld=544)
__global__ void k_filtmul(float* __restrict__ spec, const float* __restrict__ filt)
{
    const int idx = blockIdx.x * blockDim.x + threadIdx.x;
    if (idx >= 8192 * 257) return;
    const int kbin = idx % 257, frame = idx / 257;
    const float v = filt[(size_t)frame * 257 + kbin];
    float* sp = spec + (size_t)frame * 544;
    sp[kbin]       *= v;
    sp[257 + kbin] *= v;
}

// overlap-add (hop 256, win 512), keep first 32768 samples
__global__ void k_ola(const float* __restrict__ frames, float* __restrict__ ola)
{
    const int idx = blockIdx.x * blockDim.x + threadIdx.x;
    if (idx >= 64 * NRES) return;
    const int b = idx >> 15, t = idx & (NRES - 1);
    const int f1 = t >> 8;
    float acc = frames[((size_t)(b * 128 + f1)) * 512 + (t & 255)];
    if (f1 > 0) acc += frames[((size_t)(b * 128 + f1 - 1)) * 512 + (t & 255) + 256];
    ola[idx] = acc;
}

// dry/wet softmax gains
__global__ void k_fmx(const float* __restrict__ latent, const float* __restrict__ fm_w,
                      const float* __restrict__ fm_b, float* __restrict__ fmx)
{
    const int b = threadIdx.x;
    if (b >= 64) return;
    float v0 = fm_b[0], v1 = fm_b[1];
    const float* lp = latent + (size_t)b * 256;
    for (int l = 0; l < 256; ++l) { v0 += lp[l] * fm_w[l * 2]; v1 += lp[l] * fm_w[l * 2 + 1]; }
    const float mx = fmaxf(v0, v1);
    const float e0 = __expf(v0 - mx), e1 = __expf(v1 - mx);
    const float s = e0 + e1;
    fmx[b * 2] = e0 / s; fmx[b * 2 + 1] = e1 / s;
}

// per-row L2 norms of ola (32768) and impulse (16384)
__global__ void k_rownorm(const float* __restrict__ ola, const float* __restrict__ imp,
                          float* __restrict__ norms)
{
    const int b = blockIdx.x;
    const int tid = threadIdx.x;
    __shared__ float red[256];
    float s = 0.f;
    for (int i = tid; i < NRES; i += 256) { const float v = ola[((size_t)b << 15) + i]; s += v * v; }
    red[tid] = s; __syncthreads();
    for (int o = 128; o > 0; o >>= 1) { if (tid < o) red[tid] += red[tid + o]; __syncthreads(); }
    if (tid == 0) norms[2 * b] = sqrtf(red[0]);
    __syncthreads();
    s = 0.f;
    for (int i = tid; i < 16384; i += 256) { const float v = imp[(size_t)b * 16384 + i]; s += v * v; }
    red[tid] = s; __syncthreads();
    for (int o = 128; o > 0; o >>= 1) { if (tid < o) red[tid] += red[tid + o]; __syncthreads(); }
    if (tid == 0) norms[2 * b + 1] = sqrtf(red[0]);
}

__global__ void k_final(const float* __restrict__ ola, const float* __restrict__ imp,
                        const float* __restrict__ norms, const float* __restrict__ fmx,
                        float* __restrict__ out)
{
    const int idx = blockIdx.x * blockDim.x + threadIdx.x;
    if (idx >= 64 * NRES) return;
    const int b = idx >> 15, t = idx & (NRES - 1);
    float r = ola[idx] / (norms[2 * b] + 1e-8f) * fmx[2 * b];
    if (t < 16384)
        r += imp[(size_t)b * 16384 + t] / (norms[2 * b + 1] + 1e-8f) * fmx[2 * b + 1];
    out[idx] = r;
}

// ===========================================================================
// Host orchestration
// ===========================================================================
extern "C" void kernel_launch(void* const* d_in, const int* in_sizes, int n_in,
                              void* d_out, int out_size, void* d_ws, size_t ws_size,
                              hipStream_t stream)
{
    (void)in_sizes; (void)n_in; (void)out_size; (void)ws_size;
    const float* latent    = (const float*)d_in[0];
    const float* impulse   = (const float*)d_in[1];
    const float* atoms     = (const float*)d_in[2];
    const float* dec_w     = (const float*)d_in[3];
    const float* dec_b     = (const float*)d_in[4];
    const float* sel_w     = (const float*)d_in[5];
    const float* sel_b     = (const float*)d_in[6];
    const float* fm_w      = (const float*)d_in[7];
    const float* fm_b      = (const float*)d_in[8];
    const float* tf_lin_w  = (const float*)d_in[9];
    const float* tf_lin_b  = (const float*)d_in[10];
    const float* tf_conv_w = (const float*)d_in[11];
    const float* tf_conv_b = (const float*)d_in[12];
    const float* tf_bn_g   = (const float*)d_in[13];
    const float* tf_bn_b   = (const float*)d_in[14];
    const float* tf_out_w  = (const float*)d_in[15];
    const float* tf_out_b  = (const float*)d_in[16];
    const float* tm_lin_w  = (const float*)d_in[17];
    const float* tm_lin_b  = (const float*)d_in[18];
    const float* tm_conv_w = (const float*)d_in[19];
    const float* tm_conv_b = (const float*)d_in[20];
    const float* tm_bn_g   = (const float*)d_in[21];
    const float* tm_bn_b   = (const float*)d_in[22];
    const float* tm_out_w  = (const float*)d_in[23];
    const float* tm_out_b  = (const float*)d_in[24];
    float* out = (float*)d_out;

    // ---- workspace layout (floats) ----
    float* ws = (float*)d_ws;
    size_t cur = 0;
    auto alloc = [&](size_t nf) { float* p = ws + cur; cur += (nf + 63) & ~(size_t)63; return p; };
    float* d128  = alloc(64 * 128);
    float* sel   = alloc((size_t)256 * 2048);
    float* xa    = alloc((size_t)64 * 128 * 128);
    float* xb    = alloc((size_t)64 * 128 * 128);
    float* filt  = alloc((size_t)64 * 128 * 257);
    float* mx128 = alloc(64 * 4 * 128);
    float* Wf    = alloc((size_t)512 * 544);
    float* Wi    = alloc((size_t)544 * 512);
    float* norms = alloc(128);
    float* fmx   = alloc(128);
    float* bufA  = alloc((size_t)320 * NFFT * 2);
    float* bufB  = alloc((size_t)320 * NFFT * 2);
    // sub-allocations inside bufA (valid after the inverse FFT retires)
    float* finalc    = bufA;
    float* framesN   = bufA + 2097152;            // 8192*512
    float* spec      = framesN + 4194304;         // 8192*544
    float* outframes = spec + (size_t)8192 * 544; // 8192*512
    float* ola       = outframes + 4194304;       // 64*32768

    const dim3 gblk(32, 4);

    // 0) DFT matrices + zero the complex FFT input
    {
        const int tot = 512 * 544 + 544 * 512;
        k_gen_dft<<<(tot + 255) / 256, 256, 0, stream>>>(Wf, Wi);
        k_zero<<<4096, 256, 0, stream>>>(bufA, (size_t)320 * NFFT * 2);
    }
    // 1) decay envelope
    k_decay<<<64, 128, 0, stream>>>(latent, dec_w, dec_b, d128);
    // 2) sel logits (4 GEMMs, bias fused) + softmax
    for (int p = 0; p < 4; ++p) {
        k_gemm<2048, 0><<<dim3(2048 / 64, 64 / 16), gblk, 0, stream>>>(
            latent, 256, sel_w + (size_t)p * 256 * 2048,
            sel + (size_t)p * 2048, 8192, sel_b + (size_t)p * 2048,
            64, 2048, 256, nullptr, nullptr);
    }
    k_softmax2048<<<256, 256, 0, stream>>>(sel);
    // 3) res = (sel @ atoms) * decay -> complex buffer rows 0..255 (re)
    k_gemm<32768, 1><<<dim3(NRES / 64, 256 / 16), gblk, 0, stream>>>(
        sel, 2048, atoms, nullptr, 0, nullptr,
        256, NRES, 2048, d128, bufA);
    // 4) impulse -> rows 256..319
    k_pack_imp<<<(64 * 16384 + 255) / 256, 256, 0, stream>>>(impulse, bufA);
    // 5) forward FFT: radix-4 Stockham, 8 stages over 320 rows, ends in bufA
    {
        float2* fA = (float2*)bufA;
        float2* fB = (float2*)bufB;
        const int grid = (int)(((long long)320 * NQ + 255) / 256);
        for (int s = 0; s < 8; ++s)
            k_fft4_stage<<<grid, 256, 0, stream>>>((s & 1) ? fB : fA, (s & 1) ? fA : fB, 320, s);
        // 6) spectral product + conjugate -> bufB rows 0..255
        k_pointwise<<<(int)(((long long)256 * NFFT + 255) / 256), 256, 0, stream>>>(fA, fB);
        // 7) inverse FFT via forward-on-conjugate (256 rows, 8 stages, ends in bufB)
        const int grid2 = (int)(((long long)256 * NQ + 255) / 256);
        for (int s = 0; s < 8; ++s)
            k_fft4_stage<<<grid2, 256, 0, stream>>>((s & 1) ? fA : fB, (s & 1) ? fB : fA, 256, s);
    }
    // 8) tf branch: latent -> filt (64,128,257)
    {
        k_gemm<1024, 0><<<dim3(1024 / 64, 64 / 16), gblk, 0, stream>>>(
            latent, 256, tf_lin_w, xa, 1024, tf_lin_b,
            64, 1024, 256, nullptr, nullptr);
        float* pa = xa; float* pb = xb;
        for (int i = 0; i < 4; ++i) {
            const int L = 8 << i;
            const int tot = 64 * 128 * 2 * L;
            k_conv_up2<<<(tot + 255) / 256, 256, 0, stream>>>(
                pa, pb, tf_conv_w + (size_t)i * 128 * 128 * 3, tf_conv_b + i * 128, L);
            k_bnorm<<<128, 256, 0, stream>>>(pb, tf_bn_g + i * 128, tf_bn_b + i * 128, 2 * L);
            float* t = pa; pa = pb; pb = t;
        }
        k_conv_out<<<(64 * 32 * 128 + 255) / 256, 256, 0, stream>>>(pa, pb, tf_out_w, tf_out_b, 32);
        k_filt<<<(64 * 128 * 257 + 255) / 256, 256, 0, stream>>>(pb, filt);
    }
    // 9) tm branch: latent -> mx128 (64,4,128)
    {
        k_gemm<1024, 0><<<dim3(1024 / 64, 64 / 16), gblk, 0, stream>>>(
            latent, 256, tm_lin_w, xa, 1024, tm_lin_b,
            64, 1024, 256, nullptr, nullptr);
        float* pa = xa; float* pb = xb;
        for (int i = 0; i < 4; ++i) {
            const int L = 8 << i;
            const int tot = 64 * 128 * 2 * L;
            k_convtrans2<<<(tot + 255) / 256, 256, 0, stream>>>(
                pa, pb, tm_conv_w + (size_t)i * 128 * 128 * 4, tm_conv_b + i * 128, L);
            k_bnorm<<<128, 256, 0, stream>>>(pb, tm_bn_g + i * 128, tm_bn_b + i * 128, 2 * L);
            float* t = pa; pa = pb; pb = t;
        }
        k_conv_out<<<(64 * 4 * 128 + 255) / 256, 256, 0, stream>>>(pa, mx128, tm_out_w, tm_out_b, 4);
    }
    // 10) mixture-weighted IFFT combine -> final_convs (64,32768)
    k_combine<<<(64 * NRES + 255) / 256, 256, 0, stream>>>((const float2*)bufB, mx128, finalc);
    // 11) windowed frames + unit norm
    k_window<<<8192, 256, 0, stream>>>(finalc, framesN);
    // 12) rFFT-512 as WMMA GEMM: (8192x512)*(512x544) -> spec
    k_gemm<544, 0><<<dim3((544 + 63) / 64, 8192 / 16), gblk, 0, stream>>>(
        framesN, 512, Wf, spec, 544, nullptr,
        8192, 544, 512, nullptr, nullptr);
    // 13) apply time-varying spectral filter
    k_filtmul<<<(8192 * 257 + 255) / 256, 256, 0, stream>>>(spec, filt);
    // 14) irFFT-512 as WMMA GEMM: (8192x544)*(544x512) -> outframes
    k_gemm<512, 0><<<dim3(512 / 64, 8192 / 16), gblk, 0, stream>>>(
        spec, 544, Wi, outframes, 512, nullptr,
        8192, 512, 544, nullptr, nullptr);
    // 15) overlap-add
    k_ola<<<(64 * NRES + 255) / 256, 256, 0, stream>>>(outframes, ola);
    // 16) final dry/wet mix with unit norms
    k_fmx<<<1, 64, 0, stream>>>(latent, fm_w, fm_b, fmx);
    k_rownorm<<<64, 256, 0, stream>>>(ola, impulse, norms);
    k_final<<<(64 * NRES + 255) / 256, 256, 0, stream>>>(ola, impulse, norms, fmx, out);
}
